// EncoderSeq_45260365365544
// MI455X (gfx1250) — matrix operand order repaired and checked by reading, not compiled, not run on hardware
//
#include <hip/hip_runtime.h>
#include <cmath>
#include <cstdint>
#include <cstddef>

// Problem dims
#define S_LEN 512
#define BATCH 32
#define DIM   128
#define HID   512
#define GATES 2048   // 4*HID
#define NOPS_ 32

typedef __bf16 bf16_t;
typedef bf16_t v16bf __attribute__((ext_vector_type(16)));
typedef bf16_t v8bf  __attribute__((ext_vector_type(8)));
typedef float  v8f   __attribute__((ext_vector_type(8)));
typedef float  v4f   __attribute__((ext_vector_type(4)));
typedef uint32_t v4u __attribute__((ext_vector_type(4)));
typedef uint32_t v8u __attribute__((ext_vector_type(8)));

union FragU { v16bf v; v8bf h[2]; };

// A-matrix 16x32 bf16 fragment (row-major source, ld in elements).
// ISA layout: lanes 0-15 -> M=lane, elems 0-7 = K k0..k0+7, elems 8-15 = K k0+16..k0+23
//             lanes 16-31 -> M=lane-16, K k0+8..15 and k0+24..31
__device__ __forceinline__ v16bf frag_a(const bf16_t* p, int ld, int row0, int k0, int lane) {
  int r  = row0 + (lane & 15);
  int kf = (lane >> 4) << 3;            // 0 or 8
  const bf16_t* q = p + (size_t)r * ld + k0 + kf;
  FragU u;
  u.h[0] = *(const v8bf*)(q);
  u.h[1] = *(const v8bf*)(q + 16);
  return u.v;
}

// B-matrix 32x16 bf16 fragment, source stored N-major [N,K] (i.e. weight rows).
// ISA layout: lanes 0-15 hold K=0..7,16..23 of column n=lane; lanes 16-31 hold K=8..15,24..31.
__device__ __forceinline__ v16bf frag_b(const bf16_t* p, int ld, int row0, int k0, int lane) {
  int r  = row0 + (lane & 15);
  int kf = (lane >> 4) << 3;            // 0 or 8
  const bf16_t* q = p + (size_t)r * ld + k0 + kf;
  FragU u;
  u.h[0] = *(const v8bf*)(q);
  u.h[1] = *(const v8bf*)(q + 16);
  return u.v;
}

__device__ __forceinline__ v8f wmma_bf16(v16bf a, v16bf b, v8f c) {
  // D = A x B + C, f32 accumulate
  return __builtin_amdgcn_wmma_f32_16x16x32_bf16(false, a, false, b, (short)0, c, false, false);
}

__device__ __forceinline__ float sigm(float x) { return 1.0f / (1.0f + __expf(-x)); }

// ---------------------------------------------------------------------------
// Tensor Data Mover: 2D tile (global -> LDS), D# built per ISA ch.8.
// group0: [1:0]=count(1), [63:32]=lds_addr, [120:64]=global_addr, [127:126]=type(2)
// group1: [17:16]=data_size, [20]=pad_en, [24:22]=pad_interval, [31:25]=pad_amount,
//         [79:48]=tensor_dim0, [111:80]=tensor_dim1, [127:112]=tile_dim0,
//         [143:128]=tile_dim1, [207:160]=tensor_dim0_stride
__device__ __forceinline__ void tdm_load_tile(uint32_t lds_addr, const void* gaddr,
                                              uint32_t flags,
                                              uint32_t td0, uint32_t td1,
                                              uint32_t tile0, uint32_t tile1,
                                              uint32_t stride0) {
  uint64_t ga = (uint64_t)(uintptr_t)gaddr;
  v4u g0;
  g0[0] = 1u;                                  // count=1, user descriptor
  g0[1] = lds_addr;
  g0[2] = (uint32_t)ga;
  g0[3] = (uint32_t)(ga >> 32) | (2u << 30);   // type=2 ("image")
  v8u g1;
  g1[0] = flags;                               // data_size | pad cfg; workgroup_mask=0
  g1[1] = (td0 & 0xFFFFu) << 16;
  g1[2] = (td0 >> 16) | ((td1 & 0xFFFFu) << 16);
  g1[3] = (td1 >> 16) | (tile0 << 16);
  g1[4] = tile1;                               // tile_dim1; tile_dim2=0
  g1[5] = stride0;                             // tensor_dim0_stride[31:0]
  g1[6] = 0u;
  g1[7] = 0u;
  asm volatile("tensor_load_to_lds %0, %1" :: "s"(g0), "s"(g1) : "memory");
}

// bf16 (data_size=1), pad every 16 DWORDs (one 64B row) by 4 DWORDs (8 bf16):
// reproduces the [128][40] padded LDS layout used by the fragment loaders.
#define TDM_BF16_PAD ((1u << 16) | (1u << 20) | (3u << 22) | (3u << 25))

// ---------------------------------------------------------------------------
// f32 -> bf16 cast (grid-stride)
__global__ void k_cast_bf16(const float* __restrict__ src, bf16_t* __restrict__ dst, int n) {
  int stride = gridDim.x * blockDim.x;
  for (int i = blockIdx.x * blockDim.x + threadIdx.x; i < n; i += stride)
    dst[i] = (bf16_t)src[i];
}

// ---------------------------------------------------------------------------
// pre = X[16384,K] @ W[2048,K]^T + bih + bhh   (bf16 WMMA, f32 accum)
// block: 256 thr (8 waves), tile 128x128; wave tile 32x64.
// A/B tiles staged by the Tensor Data Mover, double-buffered in LDS.
__global__ __launch_bounds__(256) void k_gemm_pre(
    const bf16_t* __restrict__ X, int K,
    const bf16_t* __restrict__ W,
    const float* __restrict__ bih, const float* __restrict__ bhh,
    float* __restrict__ out) {
  __shared__ __align__(16) bf16_t As[2][128][40];
  __shared__ __align__(16) bf16_t Bs[2][128][40];
  const int t    = threadIdx.x;
  const int lane = t & 31;
  const int w    = t >> 5;
  const int m0 = blockIdx.y * 128;
  const int n0 = blockIdx.x * 128;
  const int wm = (w & 3) * 32;
  const int wn = (w >> 2) * 64;
  v8f acc[2][4];
  {
    v8f z = {};
#pragma unroll
    for (int i = 0; i < 2; ++i)
#pragma unroll
      for (int g = 0; g < 4; ++g) acc[i][g] = z;
  }
  const int nk = K >> 5;
  if (w == 0) {  // wave 0 drives the TDM; EXEC-uniform per wave
    tdm_load_tile((uint32_t)(uintptr_t)&As[0][0][0], X + (size_t)m0 * K,
                  TDM_BF16_PAD, 32, 128, 32, 128, (uint32_t)K);
    tdm_load_tile((uint32_t)(uintptr_t)&Bs[0][0][0], W + (size_t)n0 * K,
                  TDM_BF16_PAD, 32, 128, 32, 128, (uint32_t)K);
    __builtin_amdgcn_s_wait_tensorcnt(0);
  }
  __syncthreads();
  for (int kt = 0; kt < nk; ++kt) {
    const int cur = kt & 1, nxt = cur ^ 1;
    if (w == 0 && kt + 1 < nk) {       // prefetch next K-slice via TDM
      const int k1 = (kt + 1) << 5;
      tdm_load_tile((uint32_t)(uintptr_t)&As[nxt][0][0],
                    X + (size_t)m0 * K + k1, TDM_BF16_PAD, 32, 128, 32, 128, (uint32_t)K);
      tdm_load_tile((uint32_t)(uintptr_t)&Bs[nxt][0][0],
                    W + (size_t)n0 * K + k1, TDM_BF16_PAD, 32, 128, 32, 128, (uint32_t)K);
    }
    v16bf a0 = frag_a(&As[cur][0][0], 40, wm,      0, lane);
    v16bf a1 = frag_a(&As[cur][0][0], 40, wm + 16, 0, lane);
#pragma unroll
    for (int tn = 0; tn < 4; ++tn) {
      v16bf bb = frag_b(&Bs[cur][0][0], 40, wn + tn * 16, 0, lane);
      acc[0][tn] = wmma_bf16(a0, bb, acc[0][tn]);
      acc[1][tn] = wmma_bf16(a1, bb, acc[1][tn]);
    }
    __syncthreads();                                   // all reads of cur done
    if (w == 0 && kt + 1 < nk) __builtin_amdgcn_s_wait_tensorcnt(0);
    __syncthreads();                                   // nxt buffer published
  }
  const int lr = lane >> 4;
  const int lc = lane & 15;
#pragma unroll
  for (int mt = 0; mt < 2; ++mt) {
#pragma unroll
    for (int tn = 0; tn < 4; ++tn) {
      int n = n0 + wn + tn * 16 + lc;
      float bias = bih[n] + bhh[n];
#pragma unroll
      for (int v = 0; v < 8; ++v) {
        int m = m0 + wm + mt * 16 + v + 8 * lr;
        out[(size_t)m * GATES + n] = acc[mt][tn][v] + bias;
      }
    }
  }
}

// ---------------------------------------------------------------------------
// Persistent LSTM recurrence: one 1024-thread workgroup per direction.
// h lives in LDS (bf16 [32][512]); gates = h @ whh^T via bf16 WMMA each step.
// Wave w owns h-slice j in [16w,16w+16): needs gate columns {g*512+16w..} for g=i,f,g,o.
__global__ __launch_bounds__(1024) void k_lstm_rec(
    const float* __restrict__ pre_f, const float* __restrict__ pre_b,
    const bf16_t* __restrict__ whh_f, const bf16_t* __restrict__ whh_b,
    bf16_t* __restrict__ out_bf,                  // layer0: [16384,1024] concat; else null
    float* __restrict__ outf_f, float* __restrict__ outf_b) { // layer1 f32 outs; else null
  __shared__ __align__(16) bf16_t hbuf[BATCH * HID];
  const int dir = blockIdx.x;
  const bool rev = (dir == 1);
  const float*  pre  = dir ? pre_b  : pre_f;
  const bf16_t* whh  = dir ? whh_b  : whh_f;
  float*        outf = dir ? outf_b : outf_f;
  const int t    = threadIdx.x;
  const int lane = t & 31;
  const int w    = t >> 5;
  for (int i = t; i < BATCH * HID; i += 1024) hbuf[i] = (bf16_t)0.0f;
  __syncthreads();
  // Architecturally a NOP when ClusterID==0; aligns cluster-wide init if clustered.
  __builtin_amdgcn_s_cluster_barrier();
  float c_reg[2][8];
#pragma unroll
  for (int mt = 0; mt < 2; ++mt)
#pragma unroll
    for (int v = 0; v < 8; ++v) c_reg[mt][v] = 0.0f;
  const int j  = w * 16 + (lane & 15);
  const int lr = lane >> 4;
#pragma unroll 1
  for (int it = 0; it < S_LEN; ++it) {
    const int s = rev ? (S_LEN - 1 - it) : it;
    v8f acc[2][4];
    {
      v8f z = {};
#pragma unroll
      for (int i = 0; i < 2; ++i)
#pragma unroll
        for (int g = 0; g < 4; ++g) acc[i][g] = z;
    }
    // hint: pull this wave's whh slice toward the caches (gfx1250 global_prefetch)
    __builtin_prefetch(whh + (size_t)(w * 16) * HID, 0, 1);
#pragma unroll 4
    for (int kt = 0; kt < 16; ++kt) {
      const int k0 = kt * 32;
      v16bf a0 = frag_a(hbuf, HID, 0,  k0, lane);
      v16bf a1 = frag_a(hbuf, HID, 16, k0, lane);
#pragma unroll
      for (int g = 0; g < 4; ++g) {
        v16bf bb = frag_b(whh, HID, g * HID + w * 16, k0, lane);
        acc[0][g] = wmma_bf16(a0, bb, acc[0][g]);
        acc[1][g] = wmma_bf16(a1, bb, acc[1][g]);
      }
    }
    __syncthreads();                 // all reads of hbuf done before overwrite
#pragma unroll
    for (int mt = 0; mt < 2; ++mt) {
#pragma unroll
      for (int v = 0; v < 8; ++v) {
        const int batch = mt * 16 + v + 8 * lr;
        const float* prow = pre + (size_t)(s * BATCH + batch) * GATES;
        float gi = acc[mt][0][v] + prow[j];
        float gf = acc[mt][1][v] + prow[HID + j];
        float gg = acc[mt][2][v] + prow[2 * HID + j];
        float go = acc[mt][3][v] + prow[3 * HID + j];
        float c  = sigm(gf) * c_reg[mt][v] + sigm(gi) * tanhf(gg);
        c_reg[mt][v] = c;
        float h = sigm(go) * tanhf(c);
        hbuf[batch * HID + j] = (bf16_t)h;
        if (out_bf) out_bf[(size_t)(s * BATCH + batch) * 1024 + dir * HID + j] = (bf16_t)h;
        if (outf)   outf[(size_t)(s * BATCH + batch) * HID + j] = h;
      }
    }
    __syncthreads();                 // h writes visible before next step's reads
  }
}

// ---------------------------------------------------------------------------
// sq[b*512+n] = ||emb1[b,n,:]||^2
__global__ void k_sq(const float* __restrict__ emb, float* __restrict__ sq) {
  int t = blockIdx.x * blockDim.x + threadIdx.x;   // 16384 threads
  int b = t >> 9, n = t & 511;
  const float* p = emb + (size_t)(n * BATCH + b) * DIM;
  float s = 0.0f;
#pragma unroll 4
  for (int d = 0; d < DIM; d += 4) {
    v4f x = *(const v4f*)(p + d);
    s += x[0] * x[0] + x[1] * x[1] + x[2] * x[2] + x[3] * x[3];
  }
  sq[t] = s;
}

// ---------------------------------------------------------------------------
// Pass 1: pd[b,n,m] = sqrt(max(sq[n]+sq[m]-2*gram, 0)); gram via bf16 WMMA.
// Each wave async-stages its 16x128 A tile into LDS (global_load_async_to_lds),
// B fragments read straight from global. Writes pd into the word_word slot.
__global__ __launch_bounds__(256) void k_gram_pd(
    const bf16_t* __restrict__ emb_bf, const float* __restrict__ sq,
    float* __restrict__ ww, float* __restrict__ partials) {
  __shared__ __align__(16) bf16_t Ash[8][16][136];   // per-wave 16x128 tile, padded
  const int b  = blockIdx.z;
  const int n0 = blockIdx.y * 64;
  const int m0 = blockIdx.x * 64;
  const int t = threadIdx.x, lane = t & 31, w = t >> 5;
  const int tr  = w & 3;           // n-tile within block
  const int tc0 = (w >> 2) * 2;    // m-tile pair within block
  const bf16_t* base = emb_bf + (size_t)b * DIM;
  const int ld = BATCH * DIM;      // 4096: stride between consecutive emb1 rows
  {
    const bf16_t* arow0 = base + (size_t)(n0 + tr * 16) * ld;
#pragma unroll
    for (int i = 0; i < 8; ++i) {
      int idx = i * 32 + lane;     // 256 x 16B segments = 16 rows x 128 bf16
      int r = idx >> 4, seg = idx & 15;
      uint32_t lds = (uint32_t)(uintptr_t)&Ash[w][r][seg * 8];
      uint64_t ga  = (uint64_t)(uintptr_t)(arow0 + (size_t)r * ld + seg * 8);
      asm volatile("global_load_async_to_lds_b128 %0, %1, off"
                   :: "v"(lds), "v"(ga) : "memory");
    }
    asm volatile("s_wait_asynccnt 0x0" ::: "memory");  // wave-private tile ready
  }
  v8f acc[2];
  { v8f z = {}; acc[0] = z; acc[1] = z; }
#pragma unroll
  for (int kt = 0; kt < 4; ++kt) {
    const int k0 = kt * 32;
    v16bf a = frag_a(&Ash[w][0][0], 136, 0, k0, lane);
#pragma unroll
    for (int tm = 0; tm < 2; ++tm) {
      // B = emb^T in math; in N-major [N,K] form that's just emb rows again.
      v16bf bb = frag_b(base, ld, m0 + (tc0 + tm) * 16, k0, lane);
      acc[tm] = wmma_bf16(a, bb, acc[tm]);
    }
  }
  float lsum = 0.0f;
  const int lr = lane >> 4, lc = lane & 15;
#pragma unroll
  for (int tm = 0; tm < 2; ++tm) {
    const int m = m0 + (tc0 + tm) * 16 + lc;
    const float sqm = sq[b * S_LEN + m];
#pragma unroll
    for (int v = 0; v < 8; ++v) {
      const int n = n0 + tr * 16 + v + 8 * lr;
      float d2 = sq[b * S_LEN + n] + sqm - 2.0f * acc[tm][v];
      float pd = d2 > 0.0f ? sqrtf(d2) : 0.0f;
      ww[(size_t)b * S_LEN * S_LEN + (size_t)n * S_LEN + m] = pd;
      lsum += pd;
    }
  }
  __shared__ float red[256];
  red[t] = lsum;
  __syncthreads();
  for (int off = 128; off > 0; off >>= 1) {
    if (t < off) red[t] += red[t + off];
    __syncthreads();
  }
  if (t == 0) {
    int gid = (blockIdx.z * gridDim.y + blockIdx.y) * gridDim.x + blockIdx.x;
    partials[gid] = red[0];
  }
}

// Deterministic second-stage reduction -> mean
__global__ void k_reduce_mean(const float* __restrict__ partials, float* __restrict__ mean) {
  __shared__ float red[256];
  float s = 0.0f;
  for (int i = threadIdx.x; i < 2048; i += 256) s += partials[i];
  red[threadIdx.x] = s;
  __syncthreads();
  for (int off = 128; off > 0; off >>= 1) {
    if (threadIdx.x < off) red[threadIdx.x] += red[threadIdx.x + off];
    __syncthreads();
  }
  if (threadIdx.x == 0)
    mean[0] = red[0] / (float)((size_t)BATCH * S_LEN * S_LEN);
}

// Pass 2: ww = sigmoid(mean - pd), thresholded (in place)
__global__ void k_ww_finish(float* __restrict__ ww, const float* __restrict__ mean, int n) {
  const float m = mean[0];
  int stride = gridDim.x * blockDim.x;
  for (int i = blockIdx.x * blockDim.x + threadIdx.x; i < n; i += stride) {
    float v = 1.0f / (1.0f + __expf(ww[i] - m));   // sigmoid(-pd + mean)
    ww[i] = (v < 0.5f) ? 0.0f : v;
  }
}

// word_operator[b,n,o] = thresholded sigmoid(emb1[b,n] . op_emb[o])
__global__ void k_word_op(const float* __restrict__ emb, const float* __restrict__ opemb,
                          float* __restrict__ wo) {
  int idx = blockIdx.x * blockDim.x + threadIdx.x;   // 524288 threads
  int o = idx & 31;
  int n = (idx >> 5) & 511;
  int b = idx >> 14;
  const float* e = emb + (size_t)(n * BATCH + b) * DIM;
  const float* p = opemb + (size_t)o * DIM;
  float s = 0.0f;
#pragma unroll 4
  for (int d = 0; d < DIM; d += 4) {
    v4f x = *(const v4f*)(e + d);
    v4f y = *(const v4f*)(p + d);
    s = fmaf(x[0], y[0], fmaf(x[1], y[1], fmaf(x[2], y[2], fmaf(x[3], y[3], s))));
  }
  float v = 1.0f / (1.0f + __expf(-s));
  wo[idx] = (v < 0.5f) ? 0.0f : v;
}

// pade = out1f + out1b (elementwise)
__global__ void k_pade(const float* __restrict__ a, const float* __restrict__ b,
                       float* __restrict__ out, int n) {
  int stride = gridDim.x * blockDim.x;
  for (int i = blockIdx.x * blockDim.x + threadIdx.x; i < n; i += stride)
    out[i] = a[i] + b[i];
}

// problem[b,j] = out1f[S-1,b,j] + out1b[0,b,j]
__global__ void k_problem(const float* __restrict__ f, const float* __restrict__ bb,
                          float* __restrict__ out) {
  int i = blockIdx.x * blockDim.x + threadIdx.x;   // 16384 threads
  out[i] = f[(size_t)(S_LEN - 1) * BATCH * HID + i] + bb[i];
}

// ---------------------------------------------------------------------------
extern "C" void kernel_launch(void* const* d_in, const int* in_sizes, int n_in,
                              void* d_out, int out_size, void* d_ws, size_t ws_size,
                              hipStream_t stream) {
  (void)in_sizes; (void)n_in; (void)out_size; (void)ws_size;
  const float* embedded = (const float*)d_in[0];
  // d_in[1] = input_lengths (all == S, pack/pad is identity)
  const float* op_emb = (const float*)d_in[2];
  const float* wih0f = (const float*)d_in[3];
  const float* whh0f = (const float*)d_in[4];
  const float* bih0f = (const float*)d_in[5];
  const float* bhh0f = (const float*)d_in[6];
  const float* wih0b = (const float*)d_in[7];
  const float* whh0b = (const float*)d_in[8];
  const float* bih0b = (const float*)d_in[9];
  const float* bhh0b = (const float*)d_in[10];
  const float* wih1f = (const float*)d_in[11];
  const float* whh1f = (const float*)d_in[12];
  const float* bih1f = (const float*)d_in[13];
  const float* bhh1f = (const float*)d_in[14];
  const float* wih1b = (const float*)d_in[15];
  const float* whh1b = (const float*)d_in[16];
  const float* bih1b = (const float*)d_in[17];
  const float* bhh1b = (const float*)d_in[18];

  char* ws = (char*)d_ws;
  size_t off = 0;
  auto alloc = [&](size_t bytes) -> void* {
    void* p = ws + off;
    off += (bytes + 255) & ~(size_t)255;
    return p;
  };
  bf16_t* emb_bf   = (bf16_t*)alloc((size_t)S_LEN * BATCH * DIM * 2);          // 4 MB
  bf16_t* wih0f_bf = (bf16_t*)alloc((size_t)GATES * DIM * 2);
  bf16_t* wih0b_bf = (bf16_t*)alloc((size_t)GATES * DIM * 2);
  bf16_t* wih1f_bf = (bf16_t*)alloc((size_t)GATES * 1024 * 2);
  bf16_t* wih1b_bf = (bf16_t*)alloc((size_t)GATES * 1024 * 2);
  bf16_t* whh0f_bf = (bf16_t*)alloc((size_t)GATES * HID * 2);
  bf16_t* whh0b_bf = (bf16_t*)alloc((size_t)GATES * HID * 2);
  bf16_t* whh1f_bf = (bf16_t*)alloc((size_t)GATES * HID * 2);
  bf16_t* whh1b_bf = (bf16_t*)alloc((size_t)GATES * HID * 2);
  bf16_t* out0_bf  = (bf16_t*)alloc((size_t)S_LEN * BATCH * 1024 * 2);         // 32 MB
  float*  preF     = (float*)alloc((size_t)S_LEN * BATCH * GATES * 4);         // 128 MB
  float*  preB     = (float*)alloc((size_t)S_LEN * BATCH * GATES * 4);         // 128 MB
  float*  out1f    = (float*)alloc((size_t)S_LEN * BATCH * HID * 4);           // 32 MB
  float*  out1b    = (float*)alloc((size_t)S_LEN * BATCH * HID * 4);           // 32 MB
  float*  sqbuf    = (float*)alloc((size_t)BATCH * S_LEN * 4);
  float*  partials = (float*)alloc(2048 * 4);
  float*  mean_ws  = (float*)alloc(256);

  float* out_pade = (float*)d_out;                          // [S,B,H]
  float* out_prob = out_pade + (size_t)S_LEN * BATCH * HID; // [B,H]
  float* out_ww   = out_prob + (size_t)BATCH * HID;         // [B,S,S]
  float* out_wo   = out_ww + (size_t)BATCH * S_LEN * S_LEN; // [B,S,NOPS]

  // 1) casts to bf16
  k_cast_bf16<<<2048, 256, 0, stream>>>(embedded, emb_bf, S_LEN * BATCH * DIM);
  k_cast_bf16<<<512, 256, 0, stream>>>(wih0f, wih0f_bf, GATES * DIM);
  k_cast_bf16<<<512, 256, 0, stream>>>(wih0b, wih0b_bf, GATES * DIM);
  k_cast_bf16<<<2048, 256, 0, stream>>>(wih1f, wih1f_bf, GATES * 1024);
  k_cast_bf16<<<2048, 256, 0, stream>>>(wih1b, wih1b_bf, GATES * 1024);
  k_cast_bf16<<<1024, 256, 0, stream>>>(whh0f, whh0f_bf, GATES * HID);
  k_cast_bf16<<<1024, 256, 0, stream>>>(whh0b, whh0b_bf, GATES * HID);
  k_cast_bf16<<<1024, 256, 0, stream>>>(whh1f, whh1f_bf, GATES * HID);
  k_cast_bf16<<<1024, 256, 0, stream>>>(whh1b, whh1b_bf, GATES * HID);

  // 2) layer-0 input projections + recurrence
  dim3 ggrid(16, 128);
  k_gemm_pre<<<ggrid, 256, 0, stream>>>(emb_bf, DIM, wih0f_bf, bih0f, bhh0f, preF);
  k_gemm_pre<<<ggrid, 256, 0, stream>>>(emb_bf, DIM, wih0b_bf, bih0b, bhh0b, preB);
  k_lstm_rec<<<2, 1024, 0, stream>>>(preF, preB, whh0f_bf, whh0b_bf,
                                     out0_bf, nullptr, nullptr);

  // 3) layer-1 input projections + recurrence
  k_gemm_pre<<<ggrid, 256, 0, stream>>>(out0_bf, 1024, wih1f_bf, bih1f, bhh1f, preF);
  k_gemm_pre<<<ggrid, 256, 0, stream>>>(out0_bf, 1024, wih1b_bf, bih1b, bhh1b, preB);
  k_lstm_rec<<<2, 1024, 0, stream>>>(preF, preB, whh1f_bf, whh1b_bf,
                                     nullptr, out1f, out1b);

  // 4) pade + problem outputs
  k_pade<<<8192, 256, 0, stream>>>(out1f, out1b, out_pade, S_LEN * BATCH * HID);
  k_problem<<<64, 256, 0, stream>>>(out1f, out1b, out_prob);

  // 5) word_word: sq -> gram/pd (pass1) -> deterministic mean -> finish (pass2)
  k_sq<<<64, 256, 0, stream>>>(embedded, sqbuf);
  dim3 gram_grid(8, 8, 32);
  k_gram_pd<<<gram_grid, 256, 0, stream>>>(emb_bf, sqbuf, out_ww, partials);
  k_reduce_mean<<<1, 256, 0, stream>>>(partials, mean_ws);
  k_ww_finish<<<8192, 256, 0, stream>>>(out_ww, mean_ws, BATCH * S_LEN * S_LEN);

  // 6) word_operator
  k_word_op<<<2048, 256, 0, stream>>>(embedded, op_emb, out_wo);
}
